// SwinTransformerBlock_8512625181038
// MI455X (gfx1250) — matrix operand order repaired and checked by reading, not compiled, not run on hardware
//
#include <hip/hip_runtime.h>
#include <hip/hip_bf16.h>

typedef _Float16 h8  __attribute__((ext_vector_type(8)));
typedef _Float16 h16 __attribute__((ext_vector_type(16)));
typedef float    f8  __attribute__((ext_vector_type(8)));

#define N_TOK   49
#define C_DIM   192
#define H_HEADS 6
#define HID_DIM 768
#define NP      64
#define EPSV    1e-5f

// f16 weight workspace layout (element offsets)
#define WQKV_OFF  0        // 576*192 = 110592
#define WPROJ_OFF 110592   // 192*192 =  36864
#define WFC1_OFF  147456   // 768*192 = 147456
#define WFC2_OFF  294912   // 192*768 = 147456

// LDS byte offsets
#define XRES_OFF   0        // 64x192 f32 = 49152
#define BUFB_OFF   49152    // 64x192 f16 = 24576
#define REGION_OFF 73728    // 98304-byte region
#define SMEM_BYTES 172032

static __device__ __forceinline__ f8 fzero() {
  f8 c;
#pragma unroll
  for (int i = 0; i < 8; ++i) c[i] = 0.f;
  return c;
}

// A-matrix fragment (16x32 f16), documented CDNA5 layout:
// lane m = l&15, half = l>>4; VGPR0..3: K = half*8 + {0..7}; VGPR4..7: K = 16 + half*8 + {0..7}
static __device__ __forceinline__ h16 load_frag_a(const _Float16* __restrict__ p,
                                                  int row0, int ld, int k0) {
  const int lane = threadIdx.x & 31;
  const int m = lane & 15, hl = lane >> 4;
  const _Float16* q = p + (size_t)(row0 + m) * ld + k0 + hl * 8;
  h8 lo = *(const h8*)q;
  h8 hi = *(const h8*)(q + 16);
  h16 r;
#pragma unroll
  for (int i = 0; i < 8; ++i) { r[i] = lo[i]; r[8 + i] = hi[i]; }
  return r;
}

// B-matrix fragment (32x16 f16), B[k][n]; memory holds W row-major as W[n][k]
// (k contiguous). half = l>>4; VGPR0..3: K = half*8 + {0..7}; VGPR4..7: K = 16 + half*8 + {0..7}
static __device__ __forceinline__ h16 load_frag_b(const _Float16* __restrict__ p,
                                                  int n0, int ld, int k0) {
  const int lane = threadIdx.x & 31;
  const int n = lane & 15, hl = lane >> 4;
  const _Float16* q = p + (size_t)(n0 + n) * ld + k0 + hl * 8;
  h8 lo = *(const h8*)q;
  h8 hi = *(const h8*)(q + 16);
  h16 r;
#pragma unroll
  for (int i = 0; i < 8; ++i) { r[i] = lo[i]; r[8 + i] = hi[i]; }
  return r;
}

static __device__ __forceinline__ f8 wmma_f16(h16 a, h16 b, f8 c) {
  // D = A(16x32) * B(32x16) + C, f32 accum
  return __builtin_amdgcn_wmma_f32_16x16x32_f16(false, a, false, b, (short)0, c, false, false);
}

static __device__ __forceinline__ void layernorm_to_f16(
    const float* __restrict__ xin, _Float16* __restrict__ dst,
    const float* __restrict__ g, const float* __restrict__ bb,
    int wave, int lane, int tid) {
  // zero pad rows 49..63 (A-matrix padding for following GEMM)
  for (int i = tid; i < (NP - N_TOK) * C_DIM; i += 256)
    dst[N_TOK * C_DIM + i] = (_Float16)0.f;
  for (int r = wave; r < N_TOK; r += 8) {
    float v[6];
    float s = 0.f, s2 = 0.f;
#pragma unroll
    for (int k = 0; k < 6; ++k) {
      v[k] = xin[r * C_DIM + lane + 32 * k];
      s += v[k]; s2 += v[k] * v[k];
    }
#pragma unroll
    for (int o = 16; o > 0; o >>= 1) {
      s  += __shfl_xor(s,  o, 32);
      s2 += __shfl_xor(s2, o, 32);
    }
    const float mu  = s * (1.f / 192.f);
    const float var = s2 * (1.f / 192.f) - mu * mu;
    const float rs  = rsqrtf(var + EPSV);
#pragma unroll
    for (int k = 0; k < 6; ++k) {
      const int c = lane + 32 * k;
      dst[r * C_DIM + c] = (_Float16)((v[k] - mu) * rs * g[c] + bb[c]);
    }
  }
}

__global__ void __launch_bounds__(256, 1)
swin_block_kernel(const float* __restrict__ x, const int* __restrict__ amask,
                  const float* __restrict__ ln1g, const float* __restrict__ ln1b,
                  const float* __restrict__ bqkv, const float* __restrict__ bproj,
                  const float* __restrict__ ln2g, const float* __restrict__ ln2b,
                  const float* __restrict__ bfc1, const float* __restrict__ bfc2,
                  const _Float16* __restrict__ w16, float* __restrict__ out) {
  extern __shared__ __align__(16) char smem[];
  float*    xres = (float*)(smem + XRES_OFF);            // 64x192 f32 residual
  _Float16* bufB = (_Float16*)(smem + BUFB_OFF);         // 64x192 f16 (xn1 / attn-out / xn2)
  char* region = smem + REGION_OFF;
  _Float16* qk  = (_Float16*)region;                     // 64x384 f16 (Q | K), stride 384
  _Float16* vT  = (_Float16*)(region + 49152);           // 192x64 f16 (V transposed)
  float*    sc  = (float*)(region + 73728);              // 64x64 f32 scores
  _Float16* pls = (_Float16*)(region + 90112);           // 64x64 f16 softmax P
  _Float16* hb  = (_Float16*)region;                     // 64x768 f16 (MLP hidden, reuse)

  const int tid  = threadIdx.x;
  // wave index is uniform across the wave: force into an SGPR so tile loops,
  // branches and base addresses lower to SALU / scalar branches.
  const int wave = __builtin_amdgcn_readfirstlane(tid >> 5);
  const int lane = tid & 31;
  const int m    = lane & 15;
  const int hl   = lane >> 4;
  const int b    = blockIdx.x;
  const size_t xbase = (size_t)b * (N_TOK * C_DIM);

  // ---- load x window (49x192 f32) ----
  {
    const float4* xs = (const float4*)(x + xbase);
    float4* xd = (float4*)xres;
    for (int i = tid; i < (N_TOK * C_DIM) / 4; i += 256) xd[i] = xs[i];
  }
  __syncthreads();

  // ---- LN1 -> bufB (f16) ----
  layernorm_to_f16(xres, bufB, ln1g, ln1b, wave, lane, tid);
  __syncthreads();

  // ---- QKV GEMM: (64x192) @ (192x576); Q,K -> qk; V -> vT (transposed) ----
  // Each wave owns one M-tile; A fragments held in registers across 18 N-tiles.
  {
    const int ti = wave >> 1;
    const int jbase = (wave & 1) * 18;
    const int row0 = ti * 16;
    h16 af[6];
#pragma unroll
    for (int kk = 0; kk < 6; ++kk) af[kk] = load_frag_a(bufB, row0, C_DIM, kk * 32);
    for (int j = 0; j < 18; ++j) {
      const int col0 = (jbase + j) * 16;
      f8 c = fzero();
#pragma unroll
      for (int kk = 0; kk < 6; ++kk)
        c = wmma_f16(af[kk], load_frag_b(w16 + WQKV_OFF, col0, C_DIM, kk * 32), c);
      const int col = col0 + m;
      const float bias = bqkv[col];
      if (col0 < 384) {  // wave-uniform (scalar) branch
#pragma unroll
        for (int r = 0; r < 8; ++r)
          qk[(row0 + hl * 8 + r) * 384 + col] = (_Float16)(c[r] + bias);
      } else {
#pragma unroll
        for (int r = 0; r < 8; ++r)
          vT[(col - 384) * NP + (row0 + hl * 8 + r)] = (_Float16)(c[r] + bias);
      }
    }
  }
  __syncthreads();

  // ---- attention, head by head ----
  const float scale = 0.17677669529663687f;  // 1/sqrt(32)
  const int* mrow = amask + (size_t)b * (N_TOK * N_TOK);
  for (int h = 0; h < H_HEADS; ++h) {
    // S = Q K^T : 4x4 tiles, K-dim = 32; Q fragment reused across 2 N-tiles
    {
      const int ti = wave >> 1;
      const int tj0 = (wave & 1) * 2;
      h16 a = load_frag_a(qk, ti * 16, 384, h * 32);  // Q rows
#pragma unroll
      for (int jj = 0; jj < 2; ++jj) {
        const int tj = tj0 + jj;
        h16 bf = load_frag_b(qk, tj * 16, 384, 192 + h * 32);  // K rows (B[k][n]=K[n][k])
        f8 c = wmma_f16(a, bf, fzero());
#pragma unroll
        for (int r = 0; r < 8; ++r)
          sc[(ti * 16 + hl * 8 + r) * 64 + tj * 16 + m] = c[r];
      }
    }
    __syncthreads();
    // masked softmax over 49 valid cols -> pls (f16)
    for (int r = wave; r < N_TOK; r += 8) {
      const int c0 = lane, c1 = lane + 32;
      float s0 = (mrow[r * N_TOK + c0] != 0) ? sc[r * 64 + c0] * scale
                                             : -__builtin_inff();
      float s1 = -__builtin_inff();
      if (c1 < N_TOK && mrow[r * N_TOK + c1] != 0) s1 = sc[r * 64 + c1] * scale;
      float mx = fmaxf(s0, s1);
#pragma unroll
      for (int o = 16; o > 0; o >>= 1) mx = fmaxf(mx, __shfl_xor(mx, o, 32));
      float e0 = 0.f, e1 = 0.f;
      if (mx > -__builtin_inff()) { e0 = __expf(s0 - mx); e1 = __expf(s1 - mx); }
      float sum = e0 + e1;
#pragma unroll
      for (int o = 16; o > 0; o >>= 1) sum += __shfl_xor(sum, o, 32);
      const float inv = (sum > 0.f) ? 1.f / sum : 0.f;
      pls[r * 64 + c0] = (_Float16)(e0 * inv);
      pls[r * 64 + c1] = (_Float16)(e1 * inv);
    }
    __syncthreads();
    // O_h = P @ V_h : 4x2 tiles, one per wave, K = 64
    {
      const int ti = wave >> 1, jd = wave & 1;
      f8 c = fzero();
#pragma unroll
      for (int kk = 0; kk < 2; ++kk) {
        h16 a  = load_frag_a(pls, ti * 16, 64, kk * 32);
        h16 bf = load_frag_b(vT, h * 32 + jd * 16, 64, kk * 32); // B[k][n]=V[k][n]=vT[n][k]
        c = wmma_f16(a, bf, c);
      }
#pragma unroll
      for (int r = 0; r < 8; ++r)
        bufB[(ti * 16 + hl * 8 + r) * C_DIM + h * 32 + jd * 16 + m] = (_Float16)c[r];
    }
  }
  __syncthreads();

  // ---- proj: (64x192)@(192x192) + bias + residual -> xres ----
  // A fragments held in registers across 6 N-tiles per wave.
  {
    const int ti = wave >> 1;
    const int jbase = (wave & 1) * 6;
    const int row0 = ti * 16;
    h16 af[6];
#pragma unroll
    for (int kk = 0; kk < 6; ++kk) af[kk] = load_frag_a(bufB, row0, C_DIM, kk * 32);
#pragma unroll
    for (int j = 0; j < 6; ++j) {
      const int col0 = (jbase + j) * 16;
      f8 c = fzero();
#pragma unroll
      for (int kk = 0; kk < 6; ++kk)
        c = wmma_f16(af[kk], load_frag_b(w16 + WPROJ_OFF, col0, C_DIM, kk * 32), c);
      const int col = col0 + m;
      const float bias = bproj[col];
#pragma unroll
      for (int r = 0; r < 8; ++r) {
        const int row = row0 + hl * 8 + r;
        xres[row * C_DIM + col] += c[r] + bias;
      }
    }
  }
  __syncthreads();

  // ---- LN2 -> bufB ----
  layernorm_to_f16(xres, bufB, ln2g, ln2b, wave, lane, tid);
  __syncthreads();

  // ---- FC1 + GELU: (64x192)@(192x768) -> hb (f16) ----
  // A fragments held in registers across 24 N-tiles per wave.
  {
    const int ti = wave >> 1;
    const int jbase = (wave & 1) * 24;
    const int row0 = ti * 16;
    h16 af[6];
#pragma unroll
    for (int kk = 0; kk < 6; ++kk) af[kk] = load_frag_a(bufB, row0, C_DIM, kk * 32);
    for (int j = 0; j < 24; ++j) {
      const int col0 = (jbase + j) * 16;
      f8 c = fzero();
#pragma unroll
      for (int kk = 0; kk < 6; ++kk)
        c = wmma_f16(af[kk], load_frag_b(w16 + WFC1_OFF, col0, C_DIM, kk * 32), c);
      const int col = col0 + m;
      const float bias = bfc1[col];
#pragma unroll
      for (int r = 0; r < 8; ++r) {
        const float v = c[r] + bias;
        const float g = 0.5f * v * (1.f + erff(v * 0.70710678118f));  // exact GELU
        hb[(row0 + hl * 8 + r) * HID_DIM + col] = (_Float16)g;
      }
    }
  }
  __syncthreads();

  // ---- FC2: (64x768)@(768x192) + bias + residual -> out ----
  // Multi-accumulator: 6 output tiles per wave; each A fragment loaded once per
  // k-step and reused across all 6 B columns.
  {
    const int ti = wave >> 1;
    const int jbase = (wave & 1) * 6;
    const int row0 = ti * 16;
    f8 c[6];
#pragma unroll
    for (int j = 0; j < 6; ++j) c[j] = fzero();
    for (int kk = 0; kk < 24; ++kk) {
      h16 a = load_frag_a(hb, row0, HID_DIM, kk * 32);
#pragma unroll
      for (int j = 0; j < 6; ++j)
        c[j] = wmma_f16(a, load_frag_b(w16 + WFC2_OFF, (jbase + j) * 16, HID_DIM, kk * 32), c[j]);
    }
#pragma unroll
    for (int j = 0; j < 6; ++j) {
      const int col = (jbase + j) * 16 + m;
      const float bias = bfc2[col];
#pragma unroll
      for (int r = 0; r < 8; ++r) {
        const int row = row0 + hl * 8 + r;
        if (row < N_TOK)
          out[xbase + row * C_DIM + col] = xres[row * C_DIM + col] + c[j][r] + bias;
      }
    }
  }
}

__global__ void cvt_f32_to_f16(const float* __restrict__ s, _Float16* __restrict__ d, int n) {
  const int i = blockIdx.x * blockDim.x + threadIdx.x;
  if (i < n) d[i] = (_Float16)s[i];
}

extern "C" void kernel_launch(void* const* d_in, const int* in_sizes, int n_in,
                              void* d_out, int out_size, void* d_ws, size_t ws_size,
                              hipStream_t stream) {
  (void)in_sizes; (void)n_in; (void)out_size; (void)ws_size;
  const float* x     = (const float*)d_in[0];
  const int*   amask = (const int*)d_in[1];
  const float* ln1g  = (const float*)d_in[2];
  const float* ln1b  = (const float*)d_in[3];
  const float* wqkv  = (const float*)d_in[4];
  const float* bqkv  = (const float*)d_in[5];
  const float* wproj = (const float*)d_in[6];
  const float* bproj = (const float*)d_in[7];
  const float* ln2g  = (const float*)d_in[8];
  const float* ln2b  = (const float*)d_in[9];
  const float* wfc1  = (const float*)d_in[10];
  const float* bfc1  = (const float*)d_in[11];
  const float* wfc2  = (const float*)d_in[12];
  const float* bfc2  = (const float*)d_in[13];
  _Float16* w16 = (_Float16*)d_ws;

  // one-time (per launch) f32 -> f16 weight conversion into workspace
  cvt_f32_to_f16<<<(110592 + 255) / 256, 256, 0, stream>>>(wqkv,  w16 + WQKV_OFF,  110592);
  cvt_f32_to_f16<<<(36864  + 255) / 256, 256, 0, stream>>>(wproj, w16 + WPROJ_OFF, 36864);
  cvt_f32_to_f16<<<(147456 + 255) / 256, 256, 0, stream>>>(wfc1,  w16 + WFC1_OFF,  147456);
  cvt_f32_to_f16<<<(147456 + 255) / 256, 256, 0, stream>>>(wfc2,  w16 + WFC2_OFF,  147456);

  swin_block_kernel<<<4096, 256, SMEM_BYTES, stream>>>(
      x, amask, ln1g, ln1b, bqkv, bproj, ln2g, ln2b, bfc1, bfc2,
      w16, (float*)d_out);
}
